// LSTM_53910429499539
// MI455X (gfx1250) — matrix operand order repaired and checked by reading, not compiled, not run on hardware
//
#include <hip/hip_runtime.h>
#include <hip/hip_bf16.h>
#include <math.h>

// ---------------------------------------------------------------------------
// Problem constants (from reference): B=1024 S=168 P=24 F=8 H=256 Q=3 L=2
// ---------------------------------------------------------------------------
#define BB   1024
#define SS   168
#define PP   24
#define FF   8
#define HH   256
#define QQ   3
#define GATE 1024              // 4*H
#define NT   64                // GATE/16 N-tiles
#define KC0  9                 // layer0: 1 padded input chunk (9 -> 32) + 8 hidden chunks
#define KC1  16                // layer1: 8 input (h0) + 8 recurrent (h1) chunks
#define BT   16                // batch tile per block
#define HSTR 264               // LDS row stride (halves) for h buffers: 256 + 8 pad
#define XSTR 32                // LDS row stride (halves) for layer0 input tile

typedef __attribute__((ext_vector_type(16))) _Float16 v16h;
typedef __attribute__((ext_vector_type(8)))  _Float16 v8h;
typedef __attribute__((ext_vector_type(8)))  float    v8f;

// ---------------------------------------------------------------------------
// Weight pre-pack: [Wih | pad | Whh] (row n, col k) -> WMMA B-fragment order.
// dst index = ((nt*KC + kc)*32 + lane)*16 + half
//   n = nt*16 + (lane&15)
//   k = kc*32 + ((lane>>4)<<3) + (half&7) + ((half>>3)<<4)  (B 32x16 layout:
//       lanes 0-15 K=0..7,16..23; lanes 16-31 K=8..15,24..31 of the chunk)
// ---------------------------------------------------------------------------
__global__ void pack_weights(const float* __restrict__ Wih,
                             const float* __restrict__ Whh,
                             int d_in, int d_in_pad, int KC,
                             _Float16* __restrict__ dst) {
    int total = NT * KC * 512;
    for (int idx = blockIdx.x * blockDim.x + threadIdx.x; idx < total;
         idx += gridDim.x * blockDim.x) {
        int half = idx & 15;
        int lane = (idx >> 4) & 31;
        int lin  = idx >> 9;             // nt*KC + kc
        int kc   = lin % KC;
        int nt   = lin / KC;
        int n    = nt * 16 + (lane & 15);
        int k    = kc * 32 + ((lane >> 4) << 3) + (half & 7) + ((half >> 3) << 4);
        float v;
        if (k < d_in)            v = Wih[(size_t)n * d_in + k];
        else if (k < d_in_pad)   v = 0.0f;                       // zero padding
        else                     v = Whh[(size_t)n * HH + (k - d_in_pad)];
        dst[idx] = (_Float16)v;
    }
}

__global__ void pack_bias(const float* e0i, const float* e0h,
                          const float* e1i, const float* e1h,
                          const float* d0i, const float* d0h,
                          const float* d1i, const float* d1h,
                          float* __restrict__ dst) {
    int idx = blockIdx.x * blockDim.x + threadIdx.x;
    if (idx >= 4 * GATE) return;
    int layer = idx >> 10, n = idx & (GATE - 1);
    float v;
    if      (layer == 0) v = e0i[n] + e0h[n];
    else if (layer == 1) v = e1i[n] + e1h[n];
    else if (layer == 2) v = d0i[n] + d0h[n];
    else                 v = d1i[n] + d1h[n];
    dst[idx] = v;
}

// ---------------------------------------------------------------------------
// WMMA fragment loaders
// A 16x32 f16 layout: lanes 0-15 -> M=lane, K=0..7 (v0-3) and 16..23 (v4-7);
//                     lanes 16-31 -> M=lane-16, K=8..15 and 24..31.
// ---------------------------------------------------------------------------
__device__ __forceinline__ v16h load_a_lds(const _Float16* base, int stride,
                                           int kbase, int lane) {
    int m  = lane & 15;
    int kb = kbase + ((lane >> 4) << 3);
    const _Float16* p = base + m * stride + kb;
    v8h lo = *(const v8h*)(p);        // K = kb .. kb+7
    v8h hi = *(const v8h*)(p + 16);   // K = kb+16 .. kb+23
    v16h a;
#pragma unroll
    for (int i = 0; i < 8; ++i) { a[i] = lo[i]; a[i + 8] = hi[i]; }
    return a;
}

__device__ __forceinline__ float sigmoidf_(float x) {
    return 1.0f / (1.0f + __expf(-x));
}

// ---------------------------------------------------------------------------
// GEMM over a contiguous range of K-chunks. The kc loop is kept rolled
// (#pragma unroll 1) so live registers stay at acc[8] + one A frag + one B
// frag and the B loads are issued right next to their consuming WMMA —
// avoids the spill-to-scratch the compiler produced when it unrolled.
// ---------------------------------------------------------------------------
__device__ __forceinline__ void gemm_chunks(v8f (&acc)[8],
                                            const _Float16* __restrict__ wt,
                                            int KC, int kc_begin, int kc_end,
                                            const _Float16* asrc, int astride,
                                            int wv, int lane) {
    const _Float16* wl = wt + (size_t)lane * 16;   // per-lane base
#pragma unroll 1
    for (int kc = kc_begin; kc < kc_end; ++kc) {
        v16h a = load_a_lds(asrc, astride, (kc - kc_begin) * 32, lane);
#pragma unroll
        for (int j = 0; j < 8; ++j) {
            int nt = wv + 8 * j;
            v16h b = *(const v16h*)(wl + ((size_t)(nt * KC + kc)) * 512);
            acc[j] = __builtin_amdgcn_wmma_f32_16x16x32_f16(
                false, a, false, b, (short)0, acc[j], false, false);
        }
    }
}

// ---------------------------------------------------------------------------
// One LSTM layer, one timestep, for this block's 16 batch rows.
//   gates = [A0 | A1] @ Wt^T + bias   (A0 = input part, A1 = recurrent h)
// Wave w owns N-tiles {w, w+8, ..., w+56}: i/f/g/o-aligned column pairs, so
// the gate nonlinearity and cell state stay in registers (cst[2] = 32 hidden
// units per wave). h_new is written to LDS (hout) for the next consumer.
// ---------------------------------------------------------------------------
__device__ __forceinline__ void layer_step(
    const _Float16* __restrict__ wt, const float* __restrict__ bias,
    int KC, int nkc0,
    const _Float16* a0src, int a0stride,
    const _Float16* a1src,
    _Float16* hout,
    v8f (&cst)[2], int wv, int lane) {

    v8f acc[8];
#pragma unroll
    for (int j = 0; j < 8; ++j)
#pragma unroll
        for (int r = 0; r < 8; ++r) acc[j][r] = 0.0f;

    gemm_chunks(acc, wt, KC, 0,    nkc0, a0src, a0stride, wv, lane); // input part
    gemm_chunks(acc, wt, KC, nkc0, KC,   a1src, HSTR,     wv, lane); // recurrent part

    // all reads of old h are done; make writes safe across waves
    __syncthreads();

    int col  = lane & 15;
    int rofs = (lane >> 4) << 3;   // C/D layout: lanes 16-31 hold M = r+8
#pragma unroll
    for (int g = 0; g < 2; ++g) {
        int ntb  = wv + 8 * g;               // i-gate tile; f/g/o at +16/+32/+48
        int hcol = ntb * 16 + col;           // hidden unit index
        float bi = bias[(ntb)      * 16 + col];
        float bf = bias[(ntb + 16) * 16 + col];
        float bg = bias[(ntb + 32) * 16 + col];
        float bo = bias[(ntb + 48) * 16 + col];
#pragma unroll
        for (int r = 0; r < 8; ++r) {
            float xi = acc[0 + g][r] + bi;
            float xf = acc[2 + g][r] + bf;
            float xg = acc[4 + g][r] + bg;
            float xo = acc[6 + g][r] + bo;
            float cn = sigmoidf_(xf) * cst[g][r] + sigmoidf_(xi) * tanhf(xg);
            float hn = sigmoidf_(xo) * tanhf(cn);
            cst[g][r] = cn;
            hout[(r + rofs) * HSTR + hcol] = (_Float16)hn;
        }
    }
}

// ---------------------------------------------------------------------------
// Persistent LSTM kernel: one block = 16 batch rows for the whole sequence.
// __launch_bounds__(256, 1): min 1 wave/EU -> full VGPR budget, no spills.
// ---------------------------------------------------------------------------
__global__ __launch_bounds__(256, 1)
void lstm_persistent(const float* __restrict__ ht,    // (B,S,1)
                     const float* __restrict__ hist,  // (B,S,F)
                     const float* __restrict__ fut,   // (B,P,F)
                     const _Float16* __restrict__ we0,
                     const _Float16* __restrict__ we1,
                     const _Float16* __restrict__ wd0,
                     const _Float16* __restrict__ wd1,
                     const float* __restrict__ bias4, // 4*GATE (enc0,enc1,dec0,dec1)
                     const float* __restrict__ qW,    // (Q,H)
                     const float* __restrict__ qb,    // (Q)
                     const float* __restrict__ combv, // (1,Q)
                     const float* __restrict__ combg, // (1,1)
                     float* __restrict__ out) {       // (B,P,Q)
    __shared__ _Float16 x0[BT * XSTR];
    __shared__ _Float16 h0[BT * HSTR];
    __shared__ _Float16 h1[BT * HSTR];
    __shared__ float    yqbuf[BT * QQ];

    const int tid  = threadIdx.x;
    const int wv   = tid >> 5;
    const int lane = tid & 31;
    const int row0 = blockIdx.x * BT;

    for (int i = tid; i < BT * XSTR; i += 256) x0[i] = (_Float16)0.0f;
    for (int i = tid; i < BT * HSTR; i += 256) { h0[i] = (_Float16)0.0f;
                                                 h1[i] = (_Float16)0.0f; }
    v8f cst0[2], cst1[2];
#pragma unroll
    for (int g = 0; g < 2; ++g)
#pragma unroll
        for (int r = 0; r < 8; ++r) { cst0[g][r] = 0.0f; cst1[g][r] = 0.0f; }

    __syncthreads();

    // ---------------- encoder ----------------
#pragma unroll 1
    for (int t = 0; t < SS; ++t) {
        for (int i = tid; i < BT * (FF + 1); i += 256) {
            int m = i / (FF + 1), c = i % (FF + 1);
            size_t row = (size_t)(row0 + m);
            float v = (c == 0) ? ht[row * SS + t]
                               : hist[(row * SS + t) * FF + (c - 1)];
            x0[m * XSTR + c] = (_Float16)v;
        }
        __syncthreads();
        layer_step(we0, bias4,        KC0, 1, x0, XSTR, h0, h0, cst0, wv, lane);
        __syncthreads();
        layer_step(we1, bias4 + GATE, KC1, 8, h0, HSTR, h1, h1, cst1, wv, lane);
        __syncthreads();
    }

    // ---------------- decoder ----------------
    // x0 still holds the last encoder input  == xy0 of the reference.
#pragma unroll 1
    for (int t = 0; t < PP; ++t) {
        layer_step(wd0, bias4 + 2 * GATE, KC0, 1, x0, XSTR, h0, h0, cst0, wv, lane);
        __syncthreads();
        layer_step(wd1, bias4 + 3 * GATE, KC1, 8, h0, HSTR, h1, h1, cst1, wv, lane);
        __syncthreads();

        // yq = h1 @ qW^T + qb   (48 threads: one (row, q) each)
        if (tid < BT * QQ) {
            int m = tid / QQ, q = tid % QQ;
            float s = qb[q];
            for (int k = 0; k < HH; ++k)
                s += (float)h1[m * HSTR + k] * qW[q * HH + k];
            yqbuf[m * QQ + q] = s;
            out[((size_t)(row0 + m) * PP + t) * QQ + q] = s;
        }
        __syncthreads();

        // y = yq @ w_comb^T, next input = [y | future[:,t,:]]
        if (tid < BT) {
            int m = tid;
            float v0 = combv[0], v1 = combv[1], v2 = combv[2];
            float inv = combg[0] * rsqrtf(v0 * v0 + v1 * v1 + v2 * v2);
            float y = (yqbuf[m * QQ + 0] * v0 + yqbuf[m * QQ + 1] * v1 +
                       yqbuf[m * QQ + 2] * v2) * inv;
            x0[m * XSTR + 0] = (_Float16)y;
            size_t row = (size_t)(row0 + m);
#pragma unroll
            for (int c = 0; c < FF; ++c)
                x0[m * XSTR + 1 + c] = (_Float16)fut[(row * PP + t) * FF + c];
        }
        __syncthreads();
    }
}

// ---------------------------------------------------------------------------
// Host launcher
// ---------------------------------------------------------------------------
extern "C" void kernel_launch(void* const* d_in, const int* in_sizes, int n_in,
                              void* d_out, int out_size, void* d_ws, size_t ws_size,
                              hipStream_t stream) {
    const float* ht    = (const float*)d_in[0];
    const float* hist  = (const float*)d_in[1];
    const float* fut   = (const float*)d_in[2];
    const float* eW0i  = (const float*)d_in[3];
    const float* eW0h  = (const float*)d_in[4];
    const float* eb0i  = (const float*)d_in[5];
    const float* eb0h  = (const float*)d_in[6];
    const float* eW1i  = (const float*)d_in[7];
    const float* eW1h  = (const float*)d_in[8];
    const float* eb1i  = (const float*)d_in[9];
    const float* eb1h  = (const float*)d_in[10];
    const float* dW0i  = (const float*)d_in[11];
    const float* dW0h  = (const float*)d_in[12];
    const float* db0i  = (const float*)d_in[13];
    const float* db0h  = (const float*)d_in[14];
    const float* dW1i  = (const float*)d_in[15];
    const float* dW1h  = (const float*)d_in[16];
    const float* db1i  = (const float*)d_in[17];
    const float* db1h  = (const float*)d_in[18];
    const float* qW    = (const float*)d_in[19];
    const float* qb    = (const float*)d_in[20];
    const float* combv = (const float*)d_in[21];
    const float* combg = (const float*)d_in[22];
    float* out = (float*)d_out;

    const size_t sz0 = (size_t)NT * KC0 * 512;   // 294912 halves
    const size_t sz1 = (size_t)NT * KC1 * 512;   // 524288 halves
    _Float16* e0 = (_Float16*)d_ws;
    _Float16* e1 = e0 + sz0;
    _Float16* d0 = e1 + sz1;
    _Float16* d1 = d0 + sz0;
    float* bias4 = (float*)(d1 + sz1);           // 4*1024 floats

    pack_weights<<<1024, 256, 0, stream>>>(eW0i, eW0h,  9, 32, KC0, e0);
    pack_weights<<<1024, 256, 0, stream>>>(eW1i, eW1h, HH, HH, KC1, e1);
    pack_weights<<<1024, 256, 0, stream>>>(dW0i, dW0h,  9, 32, KC0, d0);
    pack_weights<<<1024, 256, 0, stream>>>(dW1i, dW1h, HH, HH, KC1, d1);
    pack_bias<<<16, 256, 0, stream>>>(eb0i, eb0h, eb1i, eb1h,
                                      db0i, db0h, db1i, db1h, bias4);

    lstm_persistent<<<BB / BT, 256, 0, stream>>>(
        ht, hist, fut, e0, e1, d0, d1, bias4, qW, qb, combv, combg, out);
}
